// AxialAttention_76794015252891
// MI455X (gfx1250) — matrix operand order repaired and checked by reading, not compiled, not run on hardware
//
#include <hip/hip_runtime.h>

typedef __attribute__((ext_vector_type(16))) _Float16 v16h;
typedef __attribute__((ext_vector_type(8)))  _Float16 v8h;
typedef __attribute__((ext_vector_type(4)))  _Float16 v4h;
typedef __attribute__((ext_vector_type(8)))  float    v8f;
typedef __attribute__((ext_vector_type(4)))  float    v4f;

// Problem constants: B=2,H=8,T=128,HH=16,WW=16,C=128
#define NPROB     4096          // B*H*HH*WW independent attention problems
#define LSEQ      128           // axial length (T)
#define CCH       128           // channels
#define T_STRIDE  32768         // HH*WW*C elements between consecutive t rows
#define OUT0_ELEMS 67108864LL   // B*H*T*HH*WW*C

// LDS layout (bytes). Row strides padded to rotate banks, 16B aligned.
#define LDH    136              // half-stride for qs/ks/vt rows (272 B)
#define LDS_F  132              // float-stride for score rows   (528 B)
#define QS_OFF 0
#define KS_OFF (128 * LDH * 2)          // 34816
#define VT_OFF (2 * 128 * LDH * 2)      // 69632
#define SS_OFF (3 * 128 * LDH * 2)      // 104448
#define LDS_BYTES (SS_OFF + 128 * LDS_F * 4)  // 172032  (< 320KB WGP pool)

__global__ __launch_bounds__(256)
void AxialAttention_76794015252891_kernel(const float* __restrict__ q,
                                          const float* __restrict__ k,
                                          const float* __restrict__ v,
                                          float* __restrict__ out) {
    extern __shared__ __align__(16) char smem[];
    _Float16* qs = (_Float16*)(smem + QS_OFF);   // [128][LDH] row-major f16 Q
    _Float16* ks = (_Float16*)(smem + KS_OFF);   // [128][LDH] row-major f16 K
    _Float16* vt = (_Float16*)(smem + VT_OFF);   // [128][LDH] f16 V transposed: vt[c][t]
    float*    ss = (float*)   (smem + SS_OFF);   // [128][LDS_F] f32 scores / probs

    const int n    = blockIdx.x;
    const int tid  = threadIdx.x;
    const int lane = tid & 31;
    const int wave = tid >> 5;

    // n = ((b*H + h)*HH + hh)*WW + ww
    const int ww = n & 15;
    const int hh = (n >> 4) & 15;
    const int bh = n >> 8;                                   // b*H + h
    const long long base = (long long)bh * 4194304LL + hh * 2048 + ww * 128;
    const float* qb = q + base;
    const float* kb = k + base;
    const float* vb = v + base;
    float* ob = out + base;                                  // out[..., t, hh, ww, c]
    float* ab = out + OUT0_ELEMS + (long long)n * (LSEQ * LSEQ);  // attn[n]

    // ---------- Phase 1: load 128x128 f32 tiles, convert to f16 into LDS ----------
    // 16384 floats / (256 threads * float4) = 16 iters; fully coalesced reads.
#pragma unroll
    for (int it = 0; it < 16; ++it) {
        const int idx = it * 256 + tid;
        const int t  = idx >> 5;            // 32 float4 per 128-wide row
        const int c4 = (idx & 31) * 4;
        const long long g = (long long)t * T_STRIDE + c4;
        const float4 fq = *(const float4*)(qb + g);
        const float4 fk = *(const float4*)(kb + g);
        const float4 fv = *(const float4*)(vb + g);
        const v4h hq = { (_Float16)fq.x, (_Float16)fq.y, (_Float16)fq.z, (_Float16)fq.w };
        const v4h hk = { (_Float16)fk.x, (_Float16)fk.y, (_Float16)fk.z, (_Float16)fk.w };
        *(v4h*)(qs + t * LDH + c4) = hq;
        *(v4h*)(ks + t * LDH + c4) = hk;
        vt[(c4 + 0) * LDH + t] = (_Float16)fv.x;   // transpose V for GEMM2 B-frags
        vt[(c4 + 1) * LDH + t] = (_Float16)fv.y;
        vt[(c4 + 2) * LDH + t] = (_Float16)fv.z;
        vt[(c4 + 3) * LDH + t] = (_Float16)fv.w;
    }
    __syncthreads();

    const int lm  = lane & 15;   // A: m-within-tile   | B: n-within-tile
    const int khi = lane >> 4;   // half-wave group

    // ---------- Phase 2: S = Q K^T / sqrt(C), wave w owns 16-row strip mi=w ----------
    {
        const int mi = wave;
        // A-fragments (16x32 f16): per ISA layout, lane(khi,m) holds
        // k = e+8*khi (e<8) and k = e+8+8*khi (e>=8) -> two contiguous 16B chunks.
        v16h afrag[4];
#pragma unroll
        for (int kc = 0; kc < 4; ++kc) {
            const _Float16* rp = qs + (16 * mi + lm) * LDH + 32 * kc + 8 * khi;
            const v8h lo = *(const v8h*)(rp);
            const v8h hi = *(const v8h*)(rp + 16);
            afrag[kc] = __builtin_shufflevector(lo, hi,
                0,1,2,3,4,5,6,7,8,9,10,11,12,13,14,15);
        }
        for (int ni = 0; ni < 8; ++ni) {
            v8f acc = {};
#pragma unroll
            for (int kc = 0; kc < 4; ++kc) {
                // B (32x16): lane(khi,n) holds k = e+8*khi (e<8), k = e+8+8*khi (e>=8).
                // B[k,n] = K[16*ni+n, 32*kc+k]
                const _Float16* rp = ks + (16 * ni + lm) * LDH + 32 * kc + 8 * khi;
                const v8h b0 = *(const v8h*)(rp);
                const v8h b1 = *(const v8h*)(rp + 16);
                const v16h bfrag = __builtin_shufflevector(b0, b1,
                    0,1,2,3,4,5,6,7,8,9,10,11,12,13,14,15);
                acc = __builtin_amdgcn_wmma_f32_16x16x32_f16(
                    false, afrag[kc], false, bfrag, (short)0, acc, false, false);
            }
            // C/D layout: VGPR r, lane -> (m = r + 8*khi, n = lane&15)
#pragma unroll
            for (int r = 0; r < 8; ++r) {
                const int m  = 16 * mi + r + 8 * khi;
                const int nn = 16 * ni + lm;
                ss[m * LDS_F + nn] = acc[r] * 0.08838834764831845f; // 1/sqrt(128)
            }
        }
    }
    __syncthreads();

    // ---------- Phase 3: row-wise softmax in LDS (one thread per row) ----------
    if (tid < 128) {
        float* row = ss + tid * LDS_F;
        float mx = -3.402823466e38f;
#pragma unroll 4
        for (int j = 0; j < 128; ++j) mx = fmaxf(mx, row[j]);
        float sum = 0.0f;
#pragma unroll 4
        for (int j = 0; j < 128; ++j) {
            const float e = __expf(row[j] - mx);
            row[j] = e;
            sum += e;
        }
        const float inv = 1.0f / sum;
#pragma unroll 4
        for (int j = 0; j < 128; ++j) row[j] *= inv;
    }
    __syncthreads();

    // ---------- Phase 4: coalesced copy of attn probabilities to global ----------
#pragma unroll
    for (int it = 0; it < 16; ++it) {
        const int idx = it * 256 + tid;
        const int t  = idx >> 5;
        const int c4 = (idx & 31) * 4;
        *(float4*)(ab + t * 128 + c4) = *(const float4*)(ss + t * LDS_F + c4);
    }
    // (no barrier needed: GEMM2 below only *reads* ss, same as the copy above)

    // ---------- Phase 5: O = P V, P re-read from ss with f32->f16 convert ----------
    {
        const int mi = wave;
        v16h afrag[4];
#pragma unroll
        for (int kc = 0; kc < 4; ++kc) {
            const float* rp = ss + (16 * mi + lm) * LDS_F + 32 * kc + 8 * khi;
            const v4f f0 = *(const v4f*)(rp);
            const v4f f1 = *(const v4f*)(rp + 4);
            const v4f f2 = *(const v4f*)(rp + 16);
            const v4f f3 = *(const v4f*)(rp + 20);
            v16h a;
#pragma unroll
            for (int i = 0; i < 4; ++i) {
                a[i]      = (_Float16)f0[i];
                a[4 + i]  = (_Float16)f1[i];
                a[8 + i]  = (_Float16)f2[i];
                a[12 + i] = (_Float16)f3[i];
            }
            afrag[kc] = a;
        }
        for (int ni = 0; ni < 8; ++ni) {
            v8f acc = {};
#pragma unroll
            for (int kc = 0; kc < 4; ++kc) {
                // B[k, c] = V[t=32*kc+k, c=16*ni+n] = vt[c][t], contiguous per lane
                const _Float16* rp = vt + (16 * ni + lm) * LDH + 32 * kc + 8 * khi;
                const v8h b0 = *(const v8h*)(rp);
                const v8h b1 = *(const v8h*)(rp + 16);
                const v16h bfrag = __builtin_shufflevector(b0, b1,
                    0,1,2,3,4,5,6,7,8,9,10,11,12,13,14,15);
                acc = __builtin_amdgcn_wmma_f32_16x16x32_f16(
                    false, afrag[kc], false, bfrag, (short)0, acc, false, false);
            }
#pragma unroll
            for (int r = 0; r < 8; ++r) {
                const int m  = 16 * mi + r + 8 * khi;   // t index
                const int cc = 16 * ni + lm;            // channel
                ob[(long long)m * T_STRIDE + cc] = acc[r];
            }
        }
    }
}

extern "C" void kernel_launch(void* const* d_in, const int* in_sizes, int n_in,
                              void* d_out, int out_size, void* d_ws, size_t ws_size,
                              hipStream_t stream) {
    (void)in_sizes; (void)n_in; (void)out_size; (void)d_ws; (void)ws_size;
    const float* q = (const float*)d_in[0];
    const float* k = (const float*)d_in[1];
    const float* v = (const float*)d_in[2];
    float* out = (float*)d_out;

    // Dynamic LDS of 168 KB (> default static cap, legal in the 320 KB WGP pool)
    hipFuncSetAttribute((const void*)AxialAttention_76794015252891_kernel,
                        hipFuncAttributeMaxDynamicSharedMemorySize, LDS_BYTES);
    AxialAttention_76794015252891_kernel<<<dim3(NPROB), dim3(256), LDS_BYTES, stream>>>(
        q, k, v, out);
}
